// WanSelfAttention_53575422050831
// MI455X (gfx1250) — matrix operand order repaired and checked by reading, not compiled, not run on hardware
//
#include <hip/hip_runtime.h>

// ---------------- problem constants (from reference setup) ----------------
#define S_LEN 3200
#define C_DIM 2048
#define NH    16
#define HD    128
#define KTAB  64      // freq table width (= head_dim/2)

typedef __attribute__((ext_vector_type(16))) __bf16 v16bf;
typedef __attribute__((ext_vector_type(8)))  float  v8f;

union Frag16 { unsigned int u[8]; v16bf v; };

__device__ __forceinline__ unsigned short f2bf(float f) {
  union { float f; unsigned int u; } c; c.f = f;
  unsigned int u = c.u;
  u += 0x7FFFu + ((u >> 16) & 1u);   // round-to-nearest-even
  return (unsigned short)(u >> 16);
}

// ---------------- fp32 -> bf16 conversion ----------------
__global__ void cvt_bf16(const float* __restrict__ in,
                         unsigned short* __restrict__ out, int n) {
  int i = blockIdx.x * blockDim.x + threadIdx.x;
  int stride = gridDim.x * blockDim.x;
  for (; i < n; i += stride) out[i] = f2bf(in[i]);
}

// ---------------- bf16 WMMA GEMM: C = A @ W^T + bias ----------------
// A: M x K (row major bf16), W: N x K (row major bf16), Cout: M x N (f32)
// Block tile 128x64, 4 waves (each 32x64), LDS double buffered, 1 barrier/K-step.
#define TPAD 40
__global__ __launch_bounds__(128)
void gemm_bf16(const unsigned short* __restrict__ A,
               const unsigned short* __restrict__ W,
               const float* __restrict__ bias,
               float* __restrict__ Cout, int M, int N, int K) {
  __shared__ unsigned short As[2][128][TPAD];
  __shared__ unsigned short Bs[2][64][TPAD];
  const int tid  = threadIdx.x;
  const int lane = tid & 31;
  const int wave = tid >> 5;
  const int lm = lane & 15;
  const int hi = lane >> 4;
  const int m0 = blockIdx.y * 128;
  const int n0 = blockIdx.x * 64;
  const int brow  = tid >> 1;
  const int bhalf = (tid & 1) * 16;

  v8f acc[2][4];
#pragma unroll
  for (int am = 0; am < 2; ++am)
#pragma unroll
    for (int t = 0; t < 4; ++t)
      acc[am][t] = (v8f){0.f,0.f,0.f,0.f,0.f,0.f,0.f,0.f};

  const unsigned short* aRow = A + (size_t)(m0 + tid) * K;          // 32 bf16 per step
  const unsigned short* bRow = W + (size_t)(n0 + brow) * K + bhalf; // 16 bf16 per step

  { // prologue: stage step 0 into buffer 0
    const uint4* ga = (const uint4*)aRow;
    uint4 a0 = ga[0], a1 = ga[1], a2 = ga[2], a3 = ga[3];
    *(uint4*)&As[0][tid][0]  = a0;  *(uint4*)&As[0][tid][8]  = a1;
    *(uint4*)&As[0][tid][16] = a2;  *(uint4*)&As[0][tid][24] = a3;
    const uint4* gb = (const uint4*)bRow;
    uint4 b0 = gb[0], b1 = gb[1];
    *(uint4*)&Bs[0][brow][bhalf] = b0; *(uint4*)&Bs[0][brow][bhalf + 8] = b1;
  }
  __syncthreads();

  const int nsteps = K / 32;
  for (int s = 0; s < nsteps; ++s) {
    const int cur = s & 1, nxt = cur ^ 1;
    const bool more = (s + 1 < nsteps);
    uint4 a0{}, a1{}, a2{}, a3{}, b0{}, b1{};
    if (more) {  // issue next tile's global loads early
      const uint4* ga = (const uint4*)(aRow + (s + 1) * 32);
      a0 = ga[0]; a1 = ga[1]; a2 = ga[2]; a3 = ga[3];
      const uint4* gb = (const uint4*)(bRow + (s + 1) * 32);
      b0 = gb[0]; b1 = gb[1];
    }

    // A fragments (rows wave*32 .. wave*32+31)
    Frag16 af0, af1;
    {
      const unsigned short* ar0 = &As[cur][wave * 32 + lm][0];
      const unsigned short* ar1 = &As[cur][wave * 32 + 16 + lm][0];
      *(uint4*)&af0.u[0] = *(const uint4*)(ar0 + hi * 8);
      *(uint4*)&af0.u[4] = *(const uint4*)(ar0 + 16 + hi * 8);
      *(uint4*)&af1.u[0] = *(const uint4*)(ar1 + hi * 8);
      *(uint4*)&af1.u[4] = *(const uint4*)(ar1 + 16 + hi * 8);
    }
#pragma unroll
    for (int t = 0; t < 4; ++t) {
      Frag16 bf_;
      const unsigned short* br = &Bs[cur][t * 16 + lm][hi * 8];
      *(uint4*)&bf_.u[0] = *(const uint4*)(br);
      *(uint4*)&bf_.u[4] = *(const uint4*)(br + 16);
      acc[0][t] = __builtin_amdgcn_wmma_f32_16x16x32_bf16(
          false, af0.v, false, bf_.v, (short)0, acc[0][t], false, false);
      acc[1][t] = __builtin_amdgcn_wmma_f32_16x16x32_bf16(
          false, af1.v, false, bf_.v, (short)0, acc[1][t], false, false);
    }

    if (more) {  // store next tile; prior barrier ordered old readers of `nxt`
      *(uint4*)&As[nxt][tid][0]  = a0;  *(uint4*)&As[nxt][tid][8]  = a1;
      *(uint4*)&As[nxt][tid][16] = a2;  *(uint4*)&As[nxt][tid][24] = a3;
      *(uint4*)&Bs[nxt][brow][bhalf] = b0; *(uint4*)&Bs[nxt][brow][bhalf + 8] = b1;
      __syncthreads();
    }
  }

#pragma unroll
  for (int am = 0; am < 2; ++am)
#pragma unroll
    for (int t = 0; t < 4; ++t) {
      int gn = n0 + t * 16 + lm;
      float bv = bias[gn];
#pragma unroll
      for (int r = 0; r < 8; ++r) {
        int gm = m0 + wave * 32 + am * 16 + r + hi * 8;
        Cout[(size_t)gm * N + gn] = acc[am][t][r] + bv;
      }
    }
}

// ---------------- RMSNorm (+RoPE) + bf16 pack. mode: 0 = q/k, 1 = v ----------------
__global__ __launch_bounds__(256)
void normrope(const float* __restrict__ pre, const float* __restrict__ g,
              const float* __restrict__ cosT, const float* __restrict__ sinT,
              const int* __restrict__ fP, const int* __restrict__ hP,
              const int* __restrict__ wP,
              unsigned short* __restrict__ out, int mode) {
  __shared__ float warp_part[8];
  __shared__ float rowbuf[C_DIM];
  const int s = blockIdx.x;
  const int tid = threadIdx.x;
  const float* row = pre + (size_t)s * C_DIM;

  float vals[8];
  float ss = 0.f;
#pragma unroll
  for (int i = 0; i < 8; ++i) {
    float v = row[tid + i * 256];
    vals[i] = v;
    ss += v * v;
  }
#pragma unroll
  for (int m = 16; m >= 1; m >>= 1) ss += __shfl_xor(ss, m, 32);
  if ((tid & 31) == 0) warp_part[tid >> 5] = ss;
  __syncthreads();

  if (mode == 1) {  // v: plain convert
#pragma unroll
    for (int i = 0; i < 8; ++i)
      out[(size_t)s * C_DIM + tid + i * 256] = f2bf(vals[i]);
    return;
  }

  float tot = 0.f;
#pragma unroll
  for (int i = 0; i < 8; ++i) tot += warp_part[i];
  float rms = rsqrtf(tot * (1.0f / C_DIM) + 1e-6f);
#pragma unroll
  for (int i = 0; i < 8; ++i)
    rowbuf[tid + i * 256] = vals[i] * rms * g[tid + i * 256];
  __syncthreads();

  const int hh = *hP, ww = *wP; (void)fP;
  const int fi = s / (hh * ww);
  const int rem = s - fi * hh * ww;
  const int hi2 = rem / ww;
  const int wi  = rem - hi2 * ww;

  for (int p = tid; p < C_DIM / 2; p += 256) {
    int head = p >> 6;
    int j = p & 63;                     // s0=22, s1=21 split of 64 cols
    int trow = (j < 22) ? fi : ((j < 43) ? hi2 : wi);
    float cv = cosT[trow * KTAB + j];
    float sv = sinT[trow * KTAB + j];
    int c0 = head * HD + 2 * j;
    float x0 = rowbuf[c0], x1 = rowbuf[c0 + 1];
    out[(size_t)s * C_DIM + c0]     = f2bf(x0 * cv - x1 * sv);
    out[(size_t)s * C_DIM + c0 + 1] = f2bf(x0 * sv + x1 * cv);
  }
}

// ---------------- flash attention: per (head, 64 q rows), double-buffered K/V ----------------
__global__ __launch_bounds__(128)
void attention(const unsigned short* __restrict__ q,
               const unsigned short* __restrict__ k,
               const unsigned short* __restrict__ v,
               unsigned short* __restrict__ o) {
  __shared__ unsigned short Ks[2][32][136];   // [key][d]
  __shared__ unsigned short Vs[2][128][40];   // [d][key] (transposed)
  __shared__ unsigned short Ps[4][16][40];    // per-wave P scratch [m][key]
  const int h  = blockIdx.y;
  const int qb = blockIdx.x * 64;
  const int tid = threadIdx.x;
  const int lane = tid & 31, wave = tid >> 5;
  const int lm = lane & 15, hi = lane >> 4;
  const int qrow = qb + wave * 16 + lm;

  // K staging mapping: thread -> (seq row, 32-d segment)
  const int kseq = tid >> 2, kdseg = (tid & 3) * 32;
  // V staging mapping: thread -> (4 consecutive seq cols, 8-d segment)
  const int vgrp = tid >> 4;            // 0..7 -> seq base = vgrp*4
  const int vseq = vgrp * 4;
  const int vdseg = (tid & 15) * 8;     // 8 d values

  // resident Q fragments: 4 chunks of K=32 over d=128
  Frag16 qf[4];
  {
    const unsigned short* qr = q + (size_t)qrow * C_DIM + h * HD;
#pragma unroll
    for (int c = 0; c < 4; ++c) {
      *(uint4*)&qf[c].u[0] = *(const uint4*)(qr + c * 32 + hi * 8);
      *(uint4*)&qf[c].u[4] = *(const uint4*)(qr + c * 32 + 16 + hi * 8);
    }
  }

  v8f oacc[8];
#pragma unroll
  for (int t = 0; t < 8; ++t) oacc[t] = (v8f){0.f,0.f,0.f,0.f,0.f,0.f,0.f,0.f};
  float mrun[8], lrun[8];
#pragma unroll
  for (int r = 0; r < 8; ++r) { mrun[r] = -1e30f; lrun[r] = 0.f; }
  const float scale = 0.08838834764831845f;   // 1/sqrt(128)

  // stage helpers operate on registers, then LDS
  auto loadK = [&](int kb, uint4* kr) {
    const uint4* gk = (const uint4*)(k + (size_t)(kb * 32 + kseq) * C_DIM + h * HD + kdseg);
    kr[0] = gk[0]; kr[1] = gk[1]; kr[2] = gk[2]; kr[3] = gk[3];
  };
  auto loadV = [&](int kb, uint4* vr) {
#pragma unroll
    for (int j = 0; j < 4; ++j)
      vr[j] = *(const uint4*)(v + (size_t)(kb * 32 + vseq + j) * C_DIM + h * HD + vdseg);
  };
  auto storeK = [&](int buf, const uint4* kr) {
    *(uint4*)&Ks[buf][kseq][kdseg]      = kr[0];
    *(uint4*)&Ks[buf][kseq][kdseg + 8]  = kr[1];
    *(uint4*)&Ks[buf][kseq][kdseg + 16] = kr[2];
    *(uint4*)&Ks[buf][kseq][kdseg + 24] = kr[3];
  };
  auto storeV = [&](int buf, const uint4* vr) {
    const unsigned short* rows = (const unsigned short*)vr;  // [4 rows][8 d]
#pragma unroll
    for (int i = 0; i < 8; ++i) {
      unsigned long long p =
          (unsigned long long)rows[0 * 8 + i]
        | ((unsigned long long)rows[1 * 8 + i] << 16)
        | ((unsigned long long)rows[2 * 8 + i] << 32)
        | ((unsigned long long)rows[3 * 8 + i] << 48);
      *(unsigned long long*)&Vs[buf][vdseg + i][vseq] = p;   // transposed b64 store
    }
  };

  { // prologue
    uint4 kr[4], vr[4];
    loadK(0, kr); loadV(0, vr);
    storeK(0, kr); storeV(0, vr);
  }
  __syncthreads();

  const int nkb = S_LEN / 32;
  for (int kb = 0; kb < nkb; ++kb) {
    const int cur = kb & 1, nxt = cur ^ 1;
    const bool more = (kb + 1 < nkb);
    uint4 kr[4] = {}, vr[4] = {};
    if (more) { loadK(kb + 1, kr); loadV(kb + 1, vr); }

    // scores: two 16x16 tiles covering 32 keys, K-loop over d
    v8f s0 = (v8f){0.f,0.f,0.f,0.f,0.f,0.f,0.f,0.f};
    v8f s1 = (v8f){0.f,0.f,0.f,0.f,0.f,0.f,0.f,0.f};
#pragma unroll
    for (int c = 0; c < 4; ++c) {
      Frag16 b0, b1;
      const unsigned short* k0r = &Ks[cur][lm][c * 32 + hi * 8];
      const unsigned short* k1r = &Ks[cur][16 + lm][c * 32 + hi * 8];
      *(uint4*)&b0.u[0] = *(const uint4*)(k0r);
      *(uint4*)&b0.u[4] = *(const uint4*)(k0r + 16);
      *(uint4*)&b1.u[0] = *(const uint4*)(k1r);
      *(uint4*)&b1.u[4] = *(const uint4*)(k1r + 16);
      s0 = __builtin_amdgcn_wmma_f32_16x16x32_bf16(false, qf[c].v, false, b0.v, (short)0, s0, false, false);
      s1 = __builtin_amdgcn_wmma_f32_16x16x32_bf16(false, qf[c].v, false, b1.v, (short)0, s1, false, false);
    }

    // online softmax; row r lives in VGPR r across the 16 lanes of each half
    float alpha[8];
#pragma unroll
    for (int r = 0; r < 8; ++r) {
      float a = s0[r] * scale, b = s1[r] * scale;
      float mx = fmaxf(a, b);
      mx = fmaxf(mx, __shfl_xor(mx, 1, 32));
      mx = fmaxf(mx, __shfl_xor(mx, 2, 32));
      mx = fmaxf(mx, __shfl_xor(mx, 4, 32));
      mx = fmaxf(mx, __shfl_xor(mx, 8, 32));
      float mnew = fmaxf(mrun[r], mx);
      alpha[r] = __expf(mrun[r] - mnew);
      mrun[r] = mnew;
      float p0 = __expf(a - mnew);
      float p1 = __expf(b - mnew);
      float rs = p0 + p1;
      rs += __shfl_xor(rs, 1, 32);
      rs += __shfl_xor(rs, 2, 32);
      rs += __shfl_xor(rs, 4, 32);
      rs += __shfl_xor(rs, 8, 32);
      lrun[r] = lrun[r] * alpha[r] + rs;
      Ps[wave][r + hi * 8][lm]      = f2bf(p0);
      Ps[wave][r + hi * 8][16 + lm] = f2bf(p1);
    }
#pragma unroll
    for (int t = 0; t < 8; ++t) {
      v8f ot = oacc[t];
#pragma unroll
      for (int r = 0; r < 8; ++r) ot[r] *= alpha[r];
      oacc[t] = ot;
    }
    // intra-wave LDS RAW: P written by all lanes, read cross-lane below
    asm volatile("s_wait_dscnt 0" ::: "memory");

    Frag16 pf;
    const unsigned short* pr = &Ps[wave][lm][0];
    *(uint4*)&pf.u[0] = *(const uint4*)(pr + hi * 8);
    *(uint4*)&pf.u[4] = *(const uint4*)(pr + 16 + hi * 8);

#pragma unroll
    for (int t = 0; t < 8; ++t) {
      Frag16 bv;
      const unsigned short* vrr = &Vs[cur][t * 16 + lm][hi * 8];
      *(uint4*)&bv.u[0] = *(const uint4*)(vrr);
      *(uint4*)&bv.u[4] = *(const uint4*)(vrr + 16);
      oacc[t] = __builtin_amdgcn_wmma_f32_16x16x32_bf16(false, pf.v, false, bv.v, (short)0, oacc[t], false, false);
    }

    if (more) {
      storeK(nxt, kr); storeV(nxt, vr);
      __syncthreads();
    }
  }

  float inv[8];
#pragma unroll
  for (int r = 0; r < 8; ++r) inv[r] = 1.f / lrun[r];
#pragma unroll
  for (int t = 0; t < 8; ++t)
#pragma unroll
    for (int r = 0; r < 8; ++r)
      o[(size_t)(qb + wave * 16 + r + hi * 8) * C_DIM + h * HD + t * 16 + lm] =
          f2bf(oacc[t][r] * inv[r]);
}

// ---------------- host launch ----------------
extern "C" void kernel_launch(void* const* d_in, const int* in_sizes, int n_in,
                              void* d_out, int out_size, void* d_ws, size_t ws_size,
                              hipStream_t stream) {
  const float* x  = (const float*)d_in[0];
  const float* wq = (const float*)d_in[1];
  const float* bq = (const float*)d_in[2];
  const float* wk = (const float*)d_in[3];
  const float* bk = (const float*)d_in[4];
  const float* wv = (const float*)d_in[5];
  const float* bv = (const float*)d_in[6];
  const float* wo = (const float*)d_in[7];
  const float* bo = (const float*)d_in[8];
  const float* gq = (const float*)d_in[9];
  const float* gk = (const float*)d_in[10];
  const float* fc = (const float*)d_in[11];
  const float* fs = (const float*)d_in[12];
  const int* fSz = (const int*)d_in[13];
  const int* hSz = (const int*)d_in[14];
  const int* wSz = (const int*)d_in[15];
  float* out = (float*)d_out;

  const size_t SC = (size_t)S_LEN * C_DIM;   // 6,553,600
  const size_t CC = (size_t)C_DIM * C_DIM;   // 4,194,304

  char* ws = (char*)d_ws;
  size_t off = 0;
  auto alloc = [&](size_t bytes) {
    void* p = ws + off;
    off += (bytes + 255) & ~(size_t)255;
    return p;
  };
  unsigned short* x_bf   = (unsigned short*)alloc(SC * 2);
  unsigned short* wq_bf  = (unsigned short*)alloc(CC * 2);
  unsigned short* wk_bf  = (unsigned short*)alloc(CC * 2);
  unsigned short* wv_bf  = (unsigned short*)alloc(CC * 2);
  unsigned short* wo_bf  = (unsigned short*)alloc(CC * 2);
  float*          pre    = (float*)alloc(SC * 4);
  unsigned short* q_bf   = (unsigned short*)alloc(SC * 2);
  unsigned short* k_bf   = (unsigned short*)alloc(SC * 2);
  unsigned short* v_bf   = (unsigned short*)alloc(SC * 2);
  unsigned short* att_bf = (unsigned short*)alloc(SC * 2);
  (void)ws_size; (void)in_sizes; (void)n_in; (void)out_size;

  // 1) bf16 conversion
  cvt_bf16<<<4096, 256, 0, stream>>>(x,  x_bf,  (int)SC);
  cvt_bf16<<<4096, 256, 0, stream>>>(wq, wq_bf, (int)CC);
  cvt_bf16<<<4096, 256, 0, stream>>>(wk, wk_bf, (int)CC);
  cvt_bf16<<<4096, 256, 0, stream>>>(wv, wv_bf, (int)CC);
  cvt_bf16<<<4096, 256, 0, stream>>>(wo, wo_bf, (int)CC);

  dim3 gGemm(C_DIM / 64, S_LEN / 128);   // (32, 25)

  // 2) Q = rmsnorm(x@wq^T + bq)*gq, rope
  gemm_bf16<<<gGemm, 128, 0, stream>>>(x_bf, wq_bf, bq, pre, S_LEN, C_DIM, C_DIM);
  normrope<<<S_LEN, 256, 0, stream>>>(pre, gq, fc, fs, fSz, hSz, wSz, q_bf, 0);
  // 3) K
  gemm_bf16<<<gGemm, 128, 0, stream>>>(x_bf, wk_bf, bk, pre, S_LEN, C_DIM, C_DIM);
  normrope<<<S_LEN, 256, 0, stream>>>(pre, gk, fc, fs, fSz, hSz, wSz, k_bf, 0);
  // 4) V (plain bias + convert)
  gemm_bf16<<<gGemm, 128, 0, stream>>>(x_bf, wv_bf, bv, pre, S_LEN, C_DIM, C_DIM);
  normrope<<<S_LEN, 256, 0, stream>>>(pre, gq, fc, fs, fSz, hSz, wSz, v_bf, 1);

  // 5) flash attention
  dim3 gAtt(S_LEN / 64, NH);             // (50, 16)
  attention<<<gAtt, 128, 0, stream>>>(q_bf, k_bf, v_bf, att_bf);

  // 6) output projection -> f32 result
  gemm_bf16<<<gGemm, 128, 0, stream>>>(att_bf, wo_bf, bo, out, S_LEN, C_DIM, C_DIM);
}
